// ConnectivityGraphGenerator_16681652977986
// MI455X (gfx1250) — matrix operand order repaired and checked by reading, not compiled, not run on hardware
//
#include <hip/hip_runtime.h>
#include <hip/hip_bf16.h>

// ---------------------------------------------------------------------------
// Problem constants (from reference): B=32, N=128, CIN=256, H=512, OUT=128
// ---------------------------------------------------------------------------
#define BSZ   32
#define NN    128
#define CIN   256
#define HH    512
#define OUTD  128
#define MROWS (BSZ * NN)          // 4096

#define TEMPERATURE 0.5f
#define EPSV 1e-8f

typedef unsigned short u16;
typedef __attribute__((ext_vector_type(16))) __bf16 v16bf;
typedef __attribute__((ext_vector_type(8)))  float  v8f;

union FragAB { v16bf v; unsigned int u[8]; };

// ---------------------------------------------------------------------------
// gfx1250 async global->LDS copy (ASYNCcnt-tracked) via inline asm — portable
// across both toolchains (bypasses the clang builtin whose pointer AS types
// differ between ROCm 7.2 and amdgpu-toolchain).
//   global_load_async_to_lds_b128 vDST, vADDR[2], off
//     vDST  = raw LDS byte offset (VGPR), vADDR = 64-bit global address.
// ---------------------------------------------------------------------------
#define USE_ASYNC_LDS 1

#if USE_ASYNC_LDS
typedef __attribute__((address_space(3))) u16 lds_u16;

__device__ __forceinline__ void async_copy_b128(const u16* gsrc, u16* lds_dst) {
    unsigned off3 = (unsigned)(size_t)(lds_u16*)lds_dst;   // LDS byte offset
    asm volatile("global_load_async_to_lds_b128 %0, %1, off"
                 :: "v"(off3), "v"(gsrc)
                 : "memory");
}
__device__ __forceinline__ void wait_async_zero() {
    asm volatile("s_wait_asynccnt 0x0" ::: "memory");
}
#endif

__device__ __forceinline__ u16 f32_to_bf16(float f) {
    unsigned int u = __float_as_uint(f);
    unsigned int r = u + 0x7FFFu + ((u >> 16) & 1u);   // round-to-nearest-even
    return (u16)(r >> 16);
}

// ---------------------------------------------------------------------------
// Kernel 1: agg = (sum_j x[b,j,:] - x[b,i,:]) / (N-1), cast to bf16.
// grid = B, block = CIN (256). Coalesced along channels.
// ---------------------------------------------------------------------------
__global__ __launch_bounds__(CIN)
void agg_cast_kernel(const float* __restrict__ x, u16* __restrict__ agg) {
    const int b = blockIdx.x;
    const int c = threadIdx.x;
    const float* xb = x + (size_t)b * NN * CIN;
    float s = 0.f;
    for (int n = 0; n < NN; ++n) s += xb[n * CIN + c];
    const float inv = 1.0f / (float)(NN - 1);
    u16* ab = agg + (size_t)b * NN * CIN;
    for (int n = 0; n < NN; ++n) {
        float a = (s - xb[n * CIN + c]) * inv;
        ab[n * CIN + c] = f32_to_bf16(a);
    }
}

// ---------------------------------------------------------------------------
// Kernel 2: f32 -> bf16 cast for weight matrices.
// ---------------------------------------------------------------------------
__global__ __launch_bounds__(256)
void cast_f32_bf16_kernel(const float* __restrict__ in, u16* __restrict__ out, int n) {
    int i = blockIdx.x * 256 + threadIdx.x;
    if (i < n) out[i] = f32_to_bf16(in[i]);
}

// ---------------------------------------------------------------------------
// Kernel 3: tiled bf16 WMMA GEMM, C = epilogue(A @ W + bias).
//   A: [M,K] bf16 row-major, W: [K,Ncols] bf16 row-major.
//   RELU_BF16 ? (relu, store bf16) : (store f32).
// Block = 256 threads (8 waves). Block tile 64x64, K staged 64 at a time via
// LDS. A tile staged with GLOBAL_LOAD_ASYNC_TO_LDS_B128 (row-major ->
// row-major, 16B chunks); B tile staged transposed (sync). Each wave owns two
// adjacent 16x16 subtiles in the same subtile-row, so the A fragment is
// loaded once per wave per k-step (2 WMMAs per fragment).
//
// Fragment layouts follow CDNA5 ISA 7.12.2 (wave32):
//   A (16x32 bf16): lane L -> row M = L%16; lane-half g = L/16;
//     VGPR v<4 : K = g*8 + 2v + {0,1};  v>=4 : K = 16 + g*8 + 2(v-4) + {0,1}
//   B (32x16 bf16): lane L -> col N = L%16; VGPR v: same K mapping as A
//     (B staged transposed in LDS so each 32-bit load grabs K,K+1 contiguously)
//   C/D (16x16 f32): VGPR i -> row M = g*8 + i, col N = L%16
// ---------------------------------------------------------------------------
#define TM 64
#define TN 64
#define KC 64
#define LDSK 72   // padded K-stride in LDS (bf16 units): 144B rows (16B aligned)

template <bool RELU_BF16>
__global__ __launch_bounds__(256)
void gemm_bf16_wmma_kernel(const u16* __restrict__ A,
                           const u16* __restrict__ W,
                           const float* __restrict__ bias,
                           void* __restrict__ Cout,
                           int M, int K, int Ncols) {
    __shared__ __align__(16) u16 sA [TM][LDSK];   // sA[m][k]
    __shared__ __align__(16) u16 sBT[TN][LDSK];   // sBT[n][k]  (W transposed)

    const int tid  = threadIdx.x;
    const int wave = tid >> 5;
    const int lane = tid & 31;
    const int g    = lane >> 4;     // lane half 0/1
    const int r    = lane & 15;     // row (A) / col (B,D) within tile

    const int row0 = blockIdx.x * TM;
    const int col0 = blockIdx.y * TN;

    // hoisted block-base pointers: inner offsets fit 32-bit
    const u16* __restrict__ Abase = A + (size_t)row0 * K;
    const u16* __restrict__ Wbase = W + col0;

    const int sub0 = wave * 2;      // two subtiles, same subtile-row
    const int sm   = sub0 >> 2;     // 0..3
    const int sn0  = sub0 & 3;      // 0 or 2

    v8f acc0 = {};
    v8f acc1 = {};

    for (int k0 = 0; k0 < K; k0 += KC) {
        const u16* __restrict__ Asrc = Abase + k0;           // TM x KC tile
        const u16* __restrict__ Wsrc = Wbase + k0 * Ncols;   // KC x TN tile

#if USE_ASYNC_LDS
        // ---- A tile via async DMA: 512 x 16B transfers, 2 per thread ----
        #pragma unroll
        for (int it = 0; it < (TM * KC) / (8 * 256); ++it) {
            int e  = tid + it * 256;
            int rr = e >> 3;            // KC/8 = 8 16B-chunks per row
            int c8 = (e & 7) * 8;
            async_copy_b128(Asrc + rr * K + c8, &sA[rr][c8]);
        }
#else
        for (int e = tid; e < TM * KC; e += 256) {
            int rr = e >> 6;            // /KC
            int cc = e & (KC - 1);
            sA[rr][cc] = Asrc[rr * K + cc];
        }
#endif
        // ---- W tile transposed: sBT[n][k], global reads coalesced along N ----
        for (int e = tid; e < KC * TN; e += 256) {
            int kk = e >> 6;            // /TN
            int nn = e & (TN - 1);
            sBT[nn][kk] = Wsrc[kk * Ncols + nn];
        }
#if USE_ASYNC_LDS
        wait_async_zero();              // this wave's DMA done before barrier
#endif
        __syncthreads();

        #pragma unroll
        for (int ks = 0; ks < KC; ks += 32) {
            FragAB fa, fb0, fb1;
            const u16* arow = &sA[sm * 16 + r][0];
            #pragma unroll
            for (int v = 0; v < 8; ++v) {
                int kloc = (v < 4) ? (ks + g * 8 + 2 * v)
                                   : (ks + 16 + g * 8 + 2 * (v - 4));
                fa.u[v] = *(const unsigned int*)(arow + kloc);
            }
            const u16* b0row = &sBT[sn0 * 16 + r][0];
            const u16* b1row = &sBT[(sn0 + 1) * 16 + r][0];
            #pragma unroll
            for (int v = 0; v < 8; ++v) {
                int kloc = (v < 4) ? (ks + g * 8 + 2 * v)
                                   : (ks + 16 + g * 8 + 2 * (v - 4));
                fb0.u[v] = *(const unsigned int*)(b0row + kloc);
                fb1.u[v] = *(const unsigned int*)(b1row + kloc);
            }
            acc0 = __builtin_amdgcn_wmma_f32_16x16x32_bf16(
                       false, fa.v, false, fb0.v, (short)0, acc0, false, false);
            acc1 = __builtin_amdgcn_wmma_f32_16x16x32_bf16(
                       false, fa.v, false, fb1.v, (short)0, acc1, false, false);
        }
        __syncthreads();
    }

    // ---- epilogue: D layout -> (row g*8+i, col r) ----
    const int gc0 = col0 + sn0 * 16 + r;
    const int gc1 = col0 + (sn0 + 1) * 16 + r;
    const float bv0 = bias[gc0];
    const float bv1 = bias[gc1];
    #pragma unroll
    for (int i = 0; i < 8; ++i) {
        int gr = row0 + sm * 16 + g * 8 + i;
        float o0 = acc0[i] + bv0;
        float o1 = acc1[i] + bv1;
        if (RELU_BF16) {
            o0 = fmaxf(o0, 0.0f);
            o1 = fmaxf(o1, 0.0f);
            ((u16*)Cout)[(size_t)gr * Ncols + gc0] = f32_to_bf16(o0);
            ((u16*)Cout)[(size_t)gr * Ncols + gc1] = f32_to_bf16(o1);
        } else {
            ((float*)Cout)[(size_t)gr * Ncols + gc0] = o0;
            ((float*)Cout)[(size_t)gr * Ncols + gc1] = o1;
        }
    }
}

// ---------------------------------------------------------------------------
// Kernel 4: pairwise Gaussian score + Gumbel-softmax + hard threshold.
// grid = B*N (one block per (b,i)), block = N (=128) threads, thread j owns
// edge (i,j). mean_i / var_i cached in LDS; softmax via LDS tree reduction.
// ---------------------------------------------------------------------------
__global__ __launch_bounds__(NN)
void pairwise_kernel(const float* __restrict__ mean, const float* __restrict__ var,
                     const float* __restrict__ gu, float* __restrict__ out) {
    const int bi = blockIdx.x;          // b*N + i
    const int j  = threadIdx.x;
    const int i  = bi & (NN - 1);
    const int rowj = bi - i + j;        // b*N + j

    __shared__ float mi[OUTD];
    __shared__ float vi[OUTD];
    __shared__ float red[NN];

    mi[j] = mean[(size_t)bi * OUTD + j];   // OUTD == NN == blockDim
    vi[j] = var [(size_t)bi * OUTD + j];
    __syncthreads();

    const float* mj = mean + (size_t)rowj * OUTD;
    const float* vj = var  + (size_t)rowj * OUTD;
    float acc = 0.f;
    #pragma unroll 4
    for (int o = 0; o < OUTD; ++o) {
        float d = mi[o] - mj[o];
        float s = vi[o] + vj[o];
        acc += __expf(-(d * d) / (2.0f * s * s + EPSV));
    }
    float p = acc * (1.0f / OUTD);
    float u = gu[(size_t)bi * NN + j];
    float gmb = -__logf(-__logf(u + EPSV) + EPSV);
    float logit = (__logf(p + EPSV) + gmb) * (1.0f / TEMPERATURE);

    // softmax over j (block-wide)
    red[j] = logit;
    __syncthreads();
    for (int off = NN / 2; off > 0; off >>= 1) {
        if (j < off) red[j] = fmaxf(red[j], red[j + off]);
        __syncthreads();
    }
    float mx = red[0];
    __syncthreads();
    float ex = __expf(logit - mx);
    red[j] = ex;
    __syncthreads();
    for (int off = NN / 2; off > 0; off >>= 1) {
        if (j < off) red[j] += red[j + off];
        __syncthreads();
    }
    float sum = red[0];
    out[(size_t)bi * NN + j] = ((ex / sum) > 0.5f) ? 1.0f : 0.0f;
}

// ---------------------------------------------------------------------------
// Launcher
// Inputs: 0 x, 1 w1_mean, 2 b1_mean, 3 w1_var, 4 b1_var,
//         5 w_mean_out, 6 b_mean_out, 7 w_var_out, 8 b_var_out, 9 gumbel_u
// ---------------------------------------------------------------------------
extern "C" void kernel_launch(void* const* d_in, const int* in_sizes, int n_in,
                              void* d_out, int out_size, void* d_ws, size_t ws_size,
                              hipStream_t stream) {
    const float* x    = (const float*)d_in[0];
    const float* w1m  = (const float*)d_in[1];
    const float* b1m  = (const float*)d_in[2];
    const float* w1v  = (const float*)d_in[3];
    const float* b1v  = (const float*)d_in[4];
    const float* w2m  = (const float*)d_in[5];
    const float* b2m  = (const float*)d_in[6];
    const float* w2v  = (const float*)d_in[7];
    const float* b2v  = (const float*)d_in[8];
    const float* gu   = (const float*)d_in[9];

    // workspace carve-out
    char* ws = (char*)d_ws;
    size_t off = 0;
    auto carve = [&](size_t bytes) -> void* {
        void* p = ws + off;
        off += (bytes + 255) & ~(size_t)255;
        return p;
    };
    u16*   agg    = (u16*)  carve((size_t)MROWS * CIN * 2);   // 2 MB
    u16*   w1m_b  = (u16*)  carve((size_t)CIN * HH * 2);      // 256 KB
    u16*   w1v_b  = (u16*)  carve((size_t)CIN * HH * 2);
    u16*   w2m_b  = (u16*)  carve((size_t)HH * OUTD * 2);     // 128 KB
    u16*   w2v_b  = (u16*)  carve((size_t)HH * OUTD * 2);
    u16*   hmean  = (u16*)  carve((size_t)MROWS * HH * 2);    // 4 MB
    u16*   hvar   = (u16*)  carve((size_t)MROWS * HH * 2);
    float* meanb  = (float*)carve((size_t)MROWS * OUTD * 4);  // 2 MB
    float* varb   = (float*)carve((size_t)MROWS * OUTD * 4);

    // 1) aggregation + bf16 cast
    agg_cast_kernel<<<BSZ, CIN, 0, stream>>>(x, agg);

    // 2) weight casts
    cast_f32_bf16_kernel<<<(CIN * HH + 255) / 256, 256, 0, stream>>>(w1m, w1m_b, CIN * HH);
    cast_f32_bf16_kernel<<<(CIN * HH + 255) / 256, 256, 0, stream>>>(w1v, w1v_b, CIN * HH);
    cast_f32_bf16_kernel<<<(HH * OUTD + 255) / 256, 256, 0, stream>>>(w2m, w2m_b, HH * OUTD);
    cast_f32_bf16_kernel<<<(HH * OUTD + 255) / 256, 256, 0, stream>>>(w2v, w2v_b, HH * OUTD);

    // 3) layer-1 GEMMs: [4096,256]x[256,512] -> relu -> bf16
    dim3 g1(MROWS / TM, HH / TN);
    gemm_bf16_wmma_kernel<true><<<g1, 256, 0, stream>>>(agg, w1m_b, b1m, hmean, MROWS, CIN, HH);
    gemm_bf16_wmma_kernel<true><<<g1, 256, 0, stream>>>(agg, w1v_b, b1v, hvar,  MROWS, CIN, HH);

    // 4) layer-2 GEMMs: [4096,512]x[512,128] -> f32
    dim3 g2(MROWS / TM, OUTD / TN);
    gemm_bf16_wmma_kernel<false><<<g2, 256, 0, stream>>>(hmean, w2m_b, b2m, meanb, MROWS, HH, OUTD);
    gemm_bf16_wmma_kernel<false><<<g2, 256, 0, stream>>>(hvar,  w2v_b, b2v, varb,  MROWS, HH, OUTD);

    // 5) pairwise score + gumbel softmax + threshold
    pairwise_kernel<<<MROWS, NN, 0, stream>>>(meanb, varb, gu, (float*)d_out);
}
